// LSTMCell_2250562863750
// MI455X (gfx1250) — matrix operand order repaired and checked by reading, not compiled, not run on hardware
//
#include <hip/hip_runtime.h>

typedef __attribute__((ext_vector_type(16))) __bf16 v16bf;
typedef __attribute__((ext_vector_type(8)))  __bf16 v8bf;
typedef __attribute__((ext_vector_type(8)))  float  v8f;
typedef __attribute__((ext_vector_type(4)))  float  v4f_;  // native vector for NT loads
typedef int v4i_ __attribute__((vector_size(16)));   // matches async-LDS builtin param

#define B_    4096
#define IN_   2048
#define H_    2048
#define BM    64        // batch rows per workgroup
#define BN    32        // h-columns per workgroup (x4 gates computed)
#define PADK  40        // 32 K-elems + 8 pad (80B row stride: 16B aligned, no bank conflicts)
#define KSTEP 32        // K per WMMA step (bf16 16x16x32)
#define NSTEPS ((IN_ + H_) / KSTEP)   // 128
#define PHSTEPS (IN_ / KSTEP)         // 64 steps per phase (x-phase, then h-phase)

#if defined(__HIP_DEVICE_COMPILE__) && defined(__gfx1250__) && \
    __has_builtin(__builtin_amdgcn_global_load_async_to_lds_b128) && \
    __has_builtin(__builtin_amdgcn_s_wait_asynccnt)
#define USE_ASYNC_LDS 1
#else
#define USE_ASYNC_LDS 0
#endif

__device__ __forceinline__ void stage16B(const __bf16* g, __bf16* l) {
#if USE_ASYNC_LDS
  __builtin_amdgcn_global_load_async_to_lds_b128(
      (__attribute__((address_space(1))) v4i_*)g,
      (__attribute__((address_space(3))) v4i_*)l,
      /*imm offset*/0, /*cpol*/0);
#else
  *(v8bf*)l = *(const v8bf*)g;   // global_load_b128 + ds_store_b128
#endif
}

__global__ __launch_bounds__(256)
void cvt_f32_to_bf16(const float* __restrict__ s, __bf16* __restrict__ d) {
  size_t i = ((size_t)blockIdx.x * 256 + threadIdx.x) * 8;
  v4f_ f0 = __builtin_nontemporal_load((const v4f_*)(s + i));      // streamed once
  v4f_ f1 = __builtin_nontemporal_load((const v4f_*)(s + i + 4));
  v8bf o;
  o[0] = (__bf16)f0[0]; o[1] = (__bf16)f0[1]; o[2] = (__bf16)f0[2]; o[3] = (__bf16)f0[3];
  o[4] = (__bf16)f1[0]; o[5] = (__bf16)f1[1]; o[6] = (__bf16)f1[2]; o[7] = (__bf16)f1[3];
  *(v8bf*)(d + i) = o;   // bf16 copies ARE reused (GEMM) -> keep cacheable
}

// Fused LSTM: gates = xb@Wxb^T + hb@Whb^T (+bias in epilogue), then activations -> hy, cy.
__global__ __launch_bounds__(256)
void lstm_wmma_kernel(const __bf16* __restrict__ xb,  const __bf16* __restrict__ hb,
                      const __bf16* __restrict__ Wxb, const __bf16* __restrict__ Whb,
                      const float*  __restrict__ bx,  const float*  __restrict__ bh,
                      const float*  __restrict__ cx,  float* __restrict__ out) {
  struct Stage { __bf16 a[2][BM][PADK]; __bf16 b[2][4][BN][PADK]; };  // 30.0 KB
  union  Smem  { Stage s; float g[4][BM][BN]; };                      // union w/ 32 KB gates
  __shared__ Smem smem;

  const int m0   = blockIdx.x * BM;
  const int n0   = blockIdx.y * BN;
  const int tid  = threadIdx.x;
  const int lane = tid & 31;
  const int wave = tid >> 5;
  const int gate = wave >> 1;   // 0..3 : i, f, g, o
  const int nh   = wave & 1;    // which 16-col half of BN
  const int half = lane >> 4;   // 0/1: WMMA lane-half
  const int lrow = lane & 15;

  // -------- staging geometry, fixed per thread --------
  // A tile: BM x 32 bf16 = 256 x 16B chunks, one per thread
  const int arow = tid >> 2;             // 0..63
  const int akc  = (tid & 3) * 8;        // K sub-chunk (8 bf16 = 16B)
  // B tiles: 4 gates x BN x 32 bf16 = 512 chunks; thread does (g0,col0) and (g0+2,col0)
  const int g0   = (tid >> 2) >> 5;      // 0..1
  const int col0 = (tid >> 2) & 31;
  const int kc0  = akc;

  const size_t aoff  = (size_t)(m0 + arow) * IN_ + akc;
  const size_t woff0 = (size_t)(g0 * H_ + n0 + col0) * IN_ + kc0;
  const size_t woff2 = woff0 + (size_t)2 * H_ * IN_;   // gate g0+2, same col/kc

  const __bf16* pax  = xb  + aoff;   // per-lane base pointers, computed once
  const __bf16* pah  = hb  + aoff;
  const __bf16* pwx0 = Wxb + woff0;
  const __bf16* pwh0 = Whb + woff0;
  const __bf16* pwx2 = Wxb + woff2;
  const __bf16* pwh2 = Whb + woff2;

  auto prefetch = [&](int buf, int ns) {
    const bool ph = ns >= PHSTEPS;             // phase 0: x/Wx, phase 1: h/Wh
    const int  ko = (ns & (PHSTEPS - 1)) * KSTEP;
    const __bf16* A  = ph ? pah  : pax;
    const __bf16* W0 = ph ? pwh0 : pwx0;
    const __bf16* W2 = ph ? pwh2 : pwx2;
    stage16B(A  + ko, &smem.s.a[buf][arow][akc]);
    stage16B(W0 + ko, &smem.s.b[buf][g0    ][col0][kc0]);
    stage16B(W2 + ko, &smem.s.b[buf][g0 + 2][col0][kc0]);
  };

  v8f acc[BM / 16] = {};

  prefetch(0, 0);
  for (int step = 0; step < NSTEPS; ++step) {
    const int buf = step & 1;
#if USE_ASYNC_LDS
    __builtin_amdgcn_s_wait_asynccnt(0);
#endif
    __syncthreads();
    if (step + 1 < NSTEPS) prefetch(buf ^ 1, step + 1);

    // B fragment: lane holds column n = lrow, K chunks {8h..8h+7, 16+8h..16+8h+7}
    v16bf bfrag;
    {
      const __bf16* bp = &smem.s.b[buf][gate][nh * 16 + lrow][8 * half];
      v8bf blo = *(const v8bf*)bp;
      v8bf bhi = *(const v8bf*)(bp + 16);
      bfrag = __builtin_shufflevector(blo, bhi, 0,1,2,3,4,5,6,7,8,9,10,11,12,13,14,15);
    }
#pragma unroll
    for (int mt = 0; mt < BM / 16; ++mt) {
      // A fragment: lane holds row m = lrow, K chunks {8h..8h+7, 16+8h..16+8h+7}
      const __bf16* ap = &smem.s.a[buf][mt * 16 + lrow][0];
      v8bf alo = *(const v8bf*)(ap + 8 * half);
      v8bf ahi = *(const v8bf*)(ap + 16 + 8 * half);
      v16bf afrag = __builtin_shufflevector(alo, ahi, 0,1,2,3,4,5,6,7,8,9,10,11,12,13,14,15);
      acc[mt] = __builtin_amdgcn_wmma_f32_16x16x32_bf16(
          false, afrag, false, bfrag, (short)0, acc[mt], false, false);
    }
    __syncthreads();
  }

  // Epilogue: accumulators -> LDS (C/D layout: M = mt*16 + r + 8*half, N = lrow)
#pragma unroll
  for (int mt = 0; mt < BM / 16; ++mt)
#pragma unroll
    for (int r = 0; r < 8; ++r)
      smem.g[gate][mt * 16 + 8 * half + r][nh * 16 + lrow] = acc[mt][r];
  __syncthreads();

  // Fused activations; coalesced, non-temporal stores of hy and cy
  for (int e = tid; e < BM * BN; e += 256) {
    int m = e >> 5, n = e & 31;
    int row = m0 + m, col = n0 + n;
    float vi = smem.g[0][m][n] + bx[0 * H_ + col] + bh[0 * H_ + col];
    float vf = smem.g[1][m][n] + bx[1 * H_ + col] + bh[1 * H_ + col];
    float vg = smem.g[2][m][n] + bx[2 * H_ + col] + bh[2 * H_ + col];
    float vo = smem.g[3][m][n] + bx[3 * H_ + col] + bh[3 * H_ + col];
    float ig = 1.0f / (1.0f + __expf(-vi));
    float fg = 1.0f / (1.0f + __expf(-vf));
    float gg = tanhf(vg);
    float og = 1.0f / (1.0f + __expf(-vo));
    float c  = __builtin_nontemporal_load(&cx[(size_t)row * H_ + col]);
    float cy = fg * c + ig * gg;
    float hy = og * tanhf(cy);
    __builtin_nontemporal_store(hy, &out[(size_t)row * H_ + col]);                   // hy
    __builtin_nontemporal_store(cy, &out[(size_t)B_ * H_ + (size_t)row * H_ + col]); // cy
  }
}

extern "C" void kernel_launch(void* const* d_in, const int* in_sizes, int n_in,
                              void* d_out, int out_size, void* d_ws, size_t ws_size,
                              hipStream_t stream) {
  const float* x  = (const float*)d_in[0];
  const float* hx = (const float*)d_in[1];
  const float* cx = (const float*)d_in[2];
  const float* Wx = (const float*)d_in[3];
  const float* bx = (const float*)d_in[4];
  const float* Wh = (const float*)d_in[5];
  const float* bh = (const float*)d_in[6];
  float* out = (float*)d_out;

  // Workspace layout (bf16): xb | hb | Wxb | Whb  = 96 MB total (fits 192 MB L2)
  __bf16* ws  = (__bf16*)d_ws;
  __bf16* xb  = ws;
  __bf16* hb  = xb  + (size_t)B_ * IN_;
  __bf16* Wxb = hb  + (size_t)B_ * H_;
  __bf16* Whb = Wxb + (size_t)4 * H_ * IN_;

  cvt_f32_to_bf16<<<(B_ * IN_)     / 2048, 256, 0, stream>>>(x,  xb);
  cvt_f32_to_bf16<<<(B_ * H_)      / 2048, 256, 0, stream>>>(hx, hb);
  cvt_f32_to_bf16<<<(4 * H_ * IN_) / 2048, 256, 0, stream>>>(Wx, Wxb);
  cvt_f32_to_bf16<<<(4 * H_ * H_)  / 2048, 256, 0, stream>>>(Wh, Whb);

  dim3 grid(B_ / BM, H_ / BN);   // 64 x 64 = 4096 workgroups
  lstm_wmma_kernel<<<grid, 256, 0, stream>>>(xb, hb, Wxb, Whb, bx, bh, cx, out);
}
